// OccupancyDecoder_10479720202559
// MI455X (gfx1250) — compile-verified, matrix operand order checked by reading, not run on hardware
//
#include <hip/hip_runtime.h>
#include <hip/hip_bf16.h>
#include <math.h>

typedef __attribute__((ext_vector_type(16))) _Float16 v16h;
typedef __attribute__((ext_vector_type(8)))  float    v8f;
typedef uint32_t u32x4 __attribute__((ext_vector_type(4)));
typedef uint32_t u32x8 __attribute__((ext_vector_type(8)));

#define EPSV 1e-5f
#define LDH 136   // padded f16 row stride, K=128 operands (bank spread: 68 dwords)
#define LDK 296   // padded f16 row stride, K=288 operands (148 dwords)
#define LDA 40    // padded f16 row stride, K=32 operands (20 dwords)
#define LDB 72    // padded f16 row stride, K=64 operands (36 dwords)

struct __align__(16) SharedMem {
  _Float16 poolT[64 * LDH];      // pooled activations, transposed [n][c], f16 B-operand
  float    x[32 * 64];           // fp32 activation plane (32 ch x 56 pos, stride 64)
  float    red[256];
  float    red2[256];
  union {
    float grid[128 * 256];       // staged master_grid tile (131072 B), filled by TDM
    struct { _Float16 wA[32 * LDH]; } s4;            // channel-reduce weights f16
    struct {
      float q[4 * 56]; float k[4 * 56];
      float attn[56 * 57];                           // fp32 scores / softmax
      _Float16 vwA[32 * LDA];                        // v_w as f16 A-operand
      _Float16 xh[64 * LDA];                         // x as f16 B-operand [n][c]
      _Float16 vH[32 * LDB];                         // v as f16 A-operand [c][j]
      _Float16 attnH[64 * LDB];                      // softmax as f16 B-operand [i][j]
    } att;
    struct { _Float16 imcol[64 * LDK]; _Float16 wres[32 * LDK];
             _Float16 w1h[16 * LDK]; float y[16 * 64]; } conv;
  } u;
};

struct Params {
  const float* grid;
  const float *w_cr, *b_cr, *bnr_w, *bnr_b, *bnr_m, *bnr_v;
  const float *q_w, *q_b, *k_w, *k_b, *v_w, *v_b, *gamma;
  const float *ln_w, *ln_b;
  const float *w_res, *b_res, *bnres_w, *bnres_b, *bnres_m, *bnres_v;
  const float *w1, *b1, *bn1_w, *bn1_b, *bn1_m, *bn1_v;
  const float *w2, *b2;
};

union FragH { v16h v; _Float16 h[16]; };
union FragF { v8f  v; float    f[8];  };

// Load a 16x32 f16 fragment (A: rows=M; B stored transposed: rows=N) from LDS.
// CDNA5 16-bit operand layout: lanes 0-15 hold K=0..7,16..23; lanes 16-31 hold
// K=8..15,24..31 for row = lane%16. Pairs are contiguous -> ds_load_b128.
__device__ inline v16h load_frag_lds(const _Float16* base, int ld, int lane) {
  const int half = (lane >> 4) & 1;
  const int r    = lane & 15;
  const _Float16* row = base + r * ld + half * 8;
  FragH fr;
#pragma unroll
  for (int v = 0; v < 8; ++v) {
    const int kb = (v < 4) ? (2 * v) : (16 + 2 * (v - 4));
    fr.h[2*v]     = row[kb];
    fr.h[2*v + 1] = row[kb + 1];
  }
  return fr.v;
}

extern __shared__ char smem_raw[];

__global__ void __launch_bounds__(256) fused_decoder(Params P, float* __restrict__ out) {
  SharedMem& S = *reinterpret_cast<SharedMem*>(smem_raw);
  const int tid  = threadIdx.x;
  const int lane = tid & 31;
  const int wave = tid >> 5;
  const int b    = blockIdx.x;

  // ---- S1: TDM DMA of master_grid[b] (131072 B) straight into LDS -----------
  // D# group0: count=1 | lds_addr | global_addr[56:0] | type=2
  // D# group1: data_size=4B, one row: tile_dim0 = tensor_dim0 = 32768 elements
  if (wave == 0) {
    const uint64_t ga  = (uint64_t)(uintptr_t)(P.grid) + (uint64_t)b * (128u * 256u * 4u);
    const uint32_t lds = (uint32_t)(uintptr_t)(void*)S.u.grid;
    u32x4 g0;
    g0[0] = 1u;                                            // count=1 (valid user D#)
    g0[1] = lds;                                           // lds_addr
    g0[2] = (uint32_t)ga;                                  // global_addr[31:0]
    g0[3] = (uint32_t)((ga >> 32) & 0x01FFFFFFu) | 0x80000000u; // addr[56:32]|type=2
    u32x8 g1;
    g1[0] = 0x00020000u;                                   // data_size=2 (4 bytes)
    g1[1] = 0x8000u << 16;                                 // tensor_dim0[15:0]=32768
    g1[2] = 1u << 16;                                      // td0 hi=0, tensor_dim1=1
    g1[3] = 0x8000u << 16;                                 // td1 hi=0, tile_dim0=32768
    g1[4] = 1u;                                            // tile_dim1=1, tile_dim2=0
    g1[5] = 32768u;                                        // tensor_dim0_stride lo32
    g1[6] = 0u;                                            // stride hi | td1_stride lo
    g1[7] = 0u;
    asm volatile("tensor_load_to_lds %0, %1" :: "s"(g0), "s"(g1) : "memory");
  }
  __builtin_amdgcn_s_wait_tensorcnt(0);
  __syncthreads();

  // ---- S2: adaptive avg-pool 16x16 -> 7x8; store f16 transposed [n][c] -------
  for (int idx = tid; idx < 64 * 128; idx += 256) {
    const int n = idx >> 7, c = idx & 127;
    _Float16 r = (_Float16)0.f;
    if (n < 56) {
      const int p = n >> 3, q = n & 7;
      const int h0 = (p * 16) / 7;
      const int h1 = ((p + 1) * 16 + 6) / 7;        // ceil
      const float* gc = S.u.grid + c * 256;
      float s = 0.f;
      for (int hh = h0; hh < h1; ++hh) {
        const float* rowp = gc + hh * 16 + q * 2;   // width bins are exact pairs
        s += rowp[0] + rowp[1];
      }
      r = (_Float16)(s * (1.f / (float)((h1 - h0) * 2)));
    }
    S.poolT[n * LDH + c] = r;
  }
  __syncthreads();

  // ---- S3: stage w_cr as f16 A-operand (reuses grid LDS region) --------------
  for (int idx = tid; idx < 32 * 128; idx += 256)
    S.u.s4.wA[(idx >> 7) * LDH + (idx & 127)] = (_Float16)P.w_cr[idx];
  __syncthreads();

  // ---- S4: channel reduce (32x128)@(128x64) via WMMA + bias + BN + ReLU ------
  {
    const int mt = wave >> 2, nt = wave & 3;        // 8 waves = 8 tiles of 16x16
    FragF acc;
#pragma unroll
    for (int i = 0; i < 8; ++i) acc.f[i] = 0.f;
#pragma unroll
    for (int kt = 0; kt < 4; ++kt) {
      v16h a  = load_frag_lds(S.u.s4.wA + mt * 16 * LDH + kt * 32, LDH, lane);
      v16h bb = load_frag_lds(S.poolT   + nt * 16 * LDH + kt * 32, LDH, lane);
      acc.v = __builtin_amdgcn_wmma_f32_16x16x32_f16(false, a, false, bb,
                                                     (short)0, acc.v, false, false);
    }
#pragma unroll
    for (int i = 0; i < 8; ++i) {
      const int M = mt * 16 + ((lane >> 4) << 3) + i;
      const int N = nt * 16 + (lane & 15);
      const float sc = P.bnr_w[M] / sqrtf(P.bnr_v[M] + EPSV);
      const float sh = P.bnr_b[M] - P.bnr_m[M] * sc;
      float v = (acc.f[i] + P.b_cr[M]) * sc + sh;
      S.x[M * 64 + N] = fmaxf(v, 0.f);
    }
  }
  __syncthreads();

  // ---- S5a: q/k projections (d=4, VALU) + f16 operand staging ----------------
  for (int idx = tid; idx < 4 * 56; idx += 256) {
    const int d = idx / 56, n = idx % 56;
    float sq = P.q_b[d], sk = P.k_b[d];
    for (int c = 0; c < 32; ++c) {
      const float xv = S.x[c * 64 + n];
      sq += P.q_w[d * 32 + c] * xv;
      sk += P.k_w[d * 32 + c] * xv;
    }
    S.u.att.q[idx] = sq; S.u.att.k[idx] = sk;
  }
  for (int idx = tid; idx < 64 * 32; idx += 256) {   // x as B-operand [n][c]
    const int n = idx >> 5, c = idx & 31;
    S.u.att.xh[n * LDA + c] = (n < 56) ? (_Float16)S.x[c * 64 + n] : (_Float16)0.f;
  }
  for (int idx = tid; idx < 32 * 32; idx += 256)     // v_w as A-operand
    S.u.att.vwA[(idx >> 5) * LDA + (idx & 31)] = (_Float16)P.v_w[idx];
  __syncthreads();

  // ---- S5b: v projection (32x32)@(32x64) via WMMA; scores on VALU ------------
  {
    const int mt = wave >> 2, nt = wave & 3;
    FragF acc;
#pragma unroll
    for (int i = 0; i < 8; ++i) acc.f[i] = 0.f;
    {
      v16h a  = load_frag_lds(S.u.att.vwA + mt * 16 * LDA, LDA, lane);
      v16h bb = load_frag_lds(S.u.att.xh  + nt * 16 * LDA, LDA, lane);
      acc.v = __builtin_amdgcn_wmma_f32_16x16x32_f16(false, a, false, bb,
                                                     (short)0, acc.v, false, false);
    }
#pragma unroll
    for (int i = 0; i < 8; ++i) {
      const int M = mt * 16 + ((lane >> 4) << 3) + i;
      const int N = nt * 16 + (lane & 15);
      S.u.att.vH[M * LDB + N] =
          (N < 56) ? (_Float16)(acc.f[i] + P.v_b[M]) : (_Float16)0.f;
    }
  }
  for (int idx = tid; idx < 56 * 56; idx += 256) {   // scores: K=4, stays VALU
    const int i = idx / 56, j = idx % 56;
    float s = 0.f;
    for (int d = 0; d < 4; ++d) s += S.u.att.q[d * 56 + i] * S.u.att.k[d * 56 + j];
    S.u.att.attn[i * 57 + j] = s;
  }
  __syncthreads();

  // ---- S5c: row-wise softmax (fp32) ------------------------------------------
  if (tid < 56) {
    float* row = S.u.att.attn + tid * 57;
    float m = row[0];
    for (int j = 1; j < 56; ++j) m = fmaxf(m, row[j]);
    float ssum = 0.f;
    for (int j = 0; j < 56; ++j) { float e = expf(row[j] - m); row[j] = e; ssum += e; }
    const float inv = 1.f / ssum;
    for (int j = 0; j < 56; ++j) row[j] *= inv;
  }
  __syncthreads();
  for (int idx = tid; idx < 64 * 64; idx += 256) {   // softmax -> f16 B-operand
    const int i = idx >> 6, j = idx & 63;
    S.u.att.attnH[i * LDB + j] =
        (i < 56 && j < 56) ? (_Float16)S.u.att.attn[i * 57 + j] : (_Float16)0.f;
  }
  __syncthreads();

  // ---- S5d: out = v @ attn^T as (32x64)@(64x64) WMMA; x += gamma*out ---------
  {
    const float gamma = P.gamma[0];
    const int mt = wave >> 2, nt = wave & 3;
    FragF acc;
#pragma unroll
    for (int i = 0; i < 8; ++i) acc.f[i] = 0.f;
#pragma unroll
    for (int kt = 0; kt < 2; ++kt) {
      v16h a  = load_frag_lds(S.u.att.vH    + mt * 16 * LDB + kt * 32, LDB, lane);
      v16h bb = load_frag_lds(S.u.att.attnH + nt * 16 * LDB + kt * 32, LDB, lane);
      acc.v = __builtin_amdgcn_wmma_f32_16x16x32_f16(false, a, false, bb,
                                                     (short)0, acc.v, false, false);
    }
#pragma unroll
    for (int i = 0; i < 8; ++i) {
      const int M = mt * 16 + ((lane >> 4) << 3) + i;
      const int N = nt * 16 + (lane & 15);
      if (N < 56) S.x[M * 64 + N] = gamma * acc.f[i] + S.x[M * 64 + N];
    }
  }
  __syncthreads();

  // ---- S6: LayerNorm over all 32*56 elements ---------------------------------
  {
    float s = 0.f, s2 = 0.f;
    for (int idx = tid; idx < 1792; idx += 256) {
      const float v = S.x[(idx / 56) * 64 + (idx % 56)];
      s += v; s2 += v * v;
    }
    S.red[tid] = s; S.red2[tid] = s2;
    __syncthreads();
    for (int st = 128; st > 0; st >>= 1) {
      if (tid < st) { S.red[tid] += S.red[tid + st]; S.red2[tid] += S.red2[tid + st]; }
      __syncthreads();
    }
    const float mu   = S.red[0] * (1.f / 1792.f);
    const float var  = S.red2[0] * (1.f / 1792.f) - mu * mu;
    const float rstd = 1.f / sqrtf(var + EPSV);
    for (int idx = tid; idx < 1792; idx += 256) {
      const int c = idx / 56, n = idx % 56;
      S.x[c * 64 + n] = (S.x[c * 64 + n] - mu) * rstd * P.ln_w[idx] + P.ln_b[idx];
    }
  }
  __syncthreads();

  // ---- stage conv weights (f16) once; then im2col helper ---------------------
  for (int idx = tid; idx < 32 * 288; idx += 256)
    S.u.conv.wres[(idx / 288) * LDK + idx % 288] = (_Float16)P.w_res[idx];
  for (int idx = tid; idx < 16 * 288; idx += 256)
    S.u.conv.w1h[(idx / 288) * LDK + idx % 288] = (_Float16)P.w1[idx];

  auto build_im2col = [&]() {
    for (int idx = tid; idx < 64 * 288; idx += 256) {
      const int n = idx / 288, k = idx % 288;
      _Float16 r = (_Float16)0.f;
      if (n < 56) {
        const int c = k / 9, rr = k % 9;
        const int hh = (n >> 3) + rr / 3 - 1;
        const int ww = (n & 7) + rr % 3 - 1;
        if (hh >= 0 && hh < 7 && ww >= 0 && ww < 8)
          r = (_Float16)S.x[c * 64 + hh * 8 + ww];
      }
      S.u.conv.imcol[n * LDK + k] = r;
    }
  };

  // ---- S7: residual 3x3 conv as GEMM (32x288)@(288x64) + BN + ReLU + skip ----
  build_im2col();
  __syncthreads();
  {
    const int mt = wave >> 2, nt = wave & 3;
    FragF acc;
#pragma unroll
    for (int i = 0; i < 8; ++i) acc.f[i] = 0.f;
#pragma unroll
    for (int kt = 0; kt < 9; ++kt) {
      v16h a  = load_frag_lds(S.u.conv.wres  + mt * 16 * LDK + kt * 32, LDK, lane);
      v16h bb = load_frag_lds(S.u.conv.imcol + nt * 16 * LDK + kt * 32, LDK, lane);
      acc.v = __builtin_amdgcn_wmma_f32_16x16x32_f16(false, a, false, bb,
                                                     (short)0, acc.v, false, false);
    }
#pragma unroll
    for (int i = 0; i < 8; ++i) {
      const int M = mt * 16 + ((lane >> 4) << 3) + i;
      const int N = nt * 16 + (lane & 15);
      if (N < 56) {
        const float sc = P.bnres_w[M] / sqrtf(P.bnres_v[M] + EPSV);
        const float sh = P.bnres_b[M] - P.bnres_m[M] * sc;
        float v = (acc.f[i] + P.b_res[M]) * sc + sh;
        S.x[M * 64 + N] += fmaxf(v, 0.f);          // + residual
      }
    }
  }
  __syncthreads();

  // ---- S8: conv1 3x3 (32->16) as GEMM (16x288)@(288x64) + BN + ReLU ----------
  build_im2col();
  __syncthreads();
  if (wave < 4) {
    const int nt = wave;
    FragF acc;
#pragma unroll
    for (int i = 0; i < 8; ++i) acc.f[i] = 0.f;
#pragma unroll
    for (int kt = 0; kt < 9; ++kt) {
      v16h a  = load_frag_lds(S.u.conv.w1h   + kt * 32, LDK, lane);
      v16h bb = load_frag_lds(S.u.conv.imcol + nt * 16 * LDK + kt * 32, LDK, lane);
      acc.v = __builtin_amdgcn_wmma_f32_16x16x32_f16(false, a, false, bb,
                                                     (short)0, acc.v, false, false);
    }
#pragma unroll
    for (int i = 0; i < 8; ++i) {
      const int M = ((lane >> 4) << 3) + i;
      const int N = nt * 16 + (lane & 15);
      if (N < 56) {
        const float sc = P.bn1_w[M] / sqrtf(P.bn1_v[M] + EPSV);
        const float sh = P.bn1_b[M] - P.bn1_m[M] * sc;
        float v = (acc.f[i] + P.b1[M]) * sc + sh;
        S.u.conv.y[M * 64 + N] = fmaxf(v, 0.f);
      }
    }
  }
  __syncthreads();

  // ---- S9: conv2 3x3 (16->1) head, write output ------------------------------
  if (tid < 56) {
    const int p = tid >> 3, q = tid & 7;
    float accv = P.b2[0];
    for (int c = 0; c < 16; ++c)
      for (int dh = -1; dh <= 1; ++dh) {
        const int hh = p + dh;
        if (hh < 0 || hh >= 7) continue;
        for (int dw = -1; dw <= 1; ++dw) {
          const int ww = q + dw;
          if (ww < 0 || ww >= 8) continue;
          accv += P.w2[c * 9 + (dh + 1) * 3 + (dw + 1)] * S.u.conv.y[c * 64 + hh * 8 + ww];
        }
      }
    out[(size_t)b * 56 + tid] = accv;
  }
}

extern "C" void kernel_launch(void* const* d_in, const int* in_sizes, int n_in,
                              void* d_out, int out_size, void* d_ws, size_t ws_size,
                              hipStream_t stream) {
  (void)n_in; (void)out_size; (void)d_ws; (void)ws_size;
  Params P;
  P.grid    = (const float*)d_in[0];
  P.w_cr    = (const float*)d_in[1];  P.b_cr   = (const float*)d_in[2];
  P.bnr_w   = (const float*)d_in[3];  P.bnr_b  = (const float*)d_in[4];
  P.bnr_m   = (const float*)d_in[5];  P.bnr_v  = (const float*)d_in[6];
  P.q_w     = (const float*)d_in[7];  P.q_b    = (const float*)d_in[8];
  P.k_w     = (const float*)d_in[9];  P.k_b    = (const float*)d_in[10];
  P.v_w     = (const float*)d_in[11]; P.v_b    = (const float*)d_in[12];
  P.gamma   = (const float*)d_in[13];
  P.ln_w    = (const float*)d_in[14]; P.ln_b   = (const float*)d_in[15];
  P.w_res   = (const float*)d_in[16]; P.b_res  = (const float*)d_in[17];
  P.bnres_w = (const float*)d_in[18]; P.bnres_b= (const float*)d_in[19];
  P.bnres_m = (const float*)d_in[20]; P.bnres_v= (const float*)d_in[21];
  P.w1      = (const float*)d_in[22]; P.b1     = (const float*)d_in[23];
  P.bn1_w   = (const float*)d_in[24]; P.bn1_b  = (const float*)d_in[25];
  P.bn1_m   = (const float*)d_in[26]; P.bn1_v  = (const float*)d_in[27];
  P.w2      = (const float*)d_in[28]; P.b2     = (const float*)d_in[29];

  const int B = in_sizes[0] / (128 * 16 * 16);
  fused_decoder<<<dim3(B), dim3(256), sizeof(SharedMem), stream>>>(P, (float*)d_out);
}
